// SelfAttention_72593537237247
// MI455X (gfx1250) — matrix operand order repaired and checked by reading, not compiled, not run on hardware
//
#include <hip/hip_runtime.h>
#include <hip/hip_bf16.h>

// ---------------- types ----------------
typedef __attribute__((ext_vector_type(16))) __bf16  v16bf;
typedef __attribute__((ext_vector_type(8)))  float   v8f;
typedef __attribute__((ext_vector_type(4)))  unsigned int u32x4;
typedef __attribute__((ext_vector_type(8)))  int     i32x8;
typedef __attribute__((ext_vector_type(4)))  int     i32x4;

union BF16Frag { v16bf v; u32x4 u[2]; };

#define S_LEN 2048
#define HQ_N  32
#define HKV_N 8
#define DH_N  128
#define DE_N  4096

__device__ __forceinline__ unsigned short f2bf(float f) {
    unsigned int u = __float_as_uint(f);
    unsigned int r = u + 0x7FFFu + ((u >> 16) & 1u);   // round-to-nearest-even
    return (unsigned short)(r >> 16);
}

// CDNA5 async global->LDS copy (ASYNCcnt-tracked, bypasses VGPRs).
__device__ __forceinline__ void async_load_b128(unsigned lds_off, const void* gptr) {
    asm volatile("global_load_async_to_lds_b128 %0, %1, off"
                 :: "v"(lds_off), "v"((unsigned long long)(uintptr_t)gptr)
                 : "memory");
}
__device__ __forceinline__ void wait_async0() {
    asm volatile("s_wait_asynccnt 0x0" ::: "memory");
}

// CDNA5 Tensor Data Mover: one DMA for a 2-D bf16 tile (with LDS row padding).
// tile_d0 = elements per row (k), tile_d1 = rows; LDS rows are padded 64B+16B.
__device__ __forceinline__ void tdm_load_2d(
    unsigned lds_off, const void* gaddr, int tensor_d0, int tensor_d1,
    int tile_d0, int tile_d1, int stride_d0) {
    unsigned long long ga = (unsigned long long)(uintptr_t)gaddr;
    u32x4 g0;
    g0[0] = 1u;                                         // count=1 (valid user D#)
    g0[1] = lds_off;                                    // LDS byte address
    g0[2] = (unsigned)(ga & 0xFFFFFFFFu);               // global_addr[31:0]
    g0[3] = (unsigned)((ga >> 32) & 0x1FFFFFFu)         // global_addr[56:32]
          | (2u << 30);                                 // type=2 (image)
    i32x8 g1;
    g1[0] = (1 << 16)                                   // data_size = 2 bytes
          | (1 << 20)                                   // pad_enable
          | (3 << 22)                                   // pad_interval: 16 DWORDs (64B)
          | (3 << 25);                                  // pad_amount: 4 DWORDs (16B)
    g1[1] = (tensor_d0 & 0xFFFF) << 16;                 // tensor_dim0[15:0]
    g1[2] = ((tensor_d0 >> 16) & 0xFFFF)                // tensor_dim0[31:16]
          | ((tensor_d1 & 0xFFFF) << 16);               // tensor_dim1[15:0]
    g1[3] = ((tensor_d1 >> 16) & 0xFFFF)                // tensor_dim1[31:16]
          | ((tile_d0 & 0xFFFF) << 16);                 // tile_dim0
    g1[4] = (tile_d1 & 0xFFFF);                         // tile_dim1 (tile_dim2=0)
    g1[5] = stride_d0;                                  // tensor_dim0_stride[31:0]
    g1[6] = 0;                                          // stride hi / dim1_stride lo
    g1[7] = 0;
    i32x4 z4 = {0, 0, 0, 0};
    i32x8 z8 = {0, 0, 0, 0, 0, 0, 0, 0};
    __builtin_amdgcn_tensor_load_to_lds(g0, g1, z4, z4, z8, 0);
}

// ---------------- fp32 -> bf16 convert ----------------
__global__ void f32_to_bf16_kernel(const float* __restrict__ X,
                                   unsigned short* __restrict__ Y, long n) {
    long i = blockIdx.x * (long)blockDim.x + threadIdx.x;
    long stride = (long)gridDim.x * blockDim.x;
    for (; i < n; i += stride) Y[i] = f2bf(X[i]);
}

// ---------------- fp32 -> bf16 transpose: X(RxC) -> Y(CxR) ----------------
__global__ __launch_bounds__(256) void transpose_f32_to_bf16(
    const float* __restrict__ X, unsigned short* __restrict__ Y, int R, int Cn) {
    __shared__ unsigned short tile[32][33];
    const int c0 = blockIdx.x * 32, r0 = blockIdx.y * 32;
    const int tc = threadIdx.x & 31, tr = threadIdx.x >> 5;   // tr in 0..7
    #pragma unroll
    for (int i = 0; i < 4; ++i)
        tile[tr + 8 * i][tc] = f2bf(X[(size_t)(r0 + tr + 8 * i) * Cn + c0 + tc]);
    __syncthreads();
    #pragma unroll
    for (int i = 0; i < 4; ++i)
        Y[(size_t)(c0 + tr + 8 * i) * R + r0 + tc] = tile[tc][tr + 8 * i];
}

// ---------------- RoPE (+ convert to bf16) ----------------
__global__ void rope_bf16_kernel(const float* __restrict__ X,
                                 const float* __restrict__ cs,
                                 unsigned short* __restrict__ Y, int H) {
    int idx = blockIdx.x * blockDim.x + threadIdx.x;     // over S*H*64
    int total = S_LEN * H * 64;
    if (idx >= total) return;
    int d = idx & 63;
    int h = (idx >> 6) % H;
    int s = idx / (64 * H);
    float c  = cs[s * 64 + d];
    float sn = cs[S_LEN * 64 + s * 64 + d];
    size_t base = ((size_t)s * H + h) * DH_N;
    float x1 = X[base + d], x2 = X[base + 64 + d];
    Y[base + d]      = f2bf( x1 * c  + x2 * sn);
    Y[base + 64 + d] = f2bf(-x1 * sn + x2 * c );
}

// ---------------- tiled bf16 WMMA GEMM: C(MxN,f32) = A(MxK) * Bt(NxK)^T ----------------
// A row-major (staged by TDM), B pre-transposed NxK (staged by async copies).
// Double-buffered. Block 128 threads (4 waves); wave computes 32x64 (8 WMMAs/k-step).
#define TM 128
#define TN 64
#define TK 32
__global__ __launch_bounds__(128) void gemm_bf16_wmma(
    const unsigned short* __restrict__ A,
    const unsigned short* __restrict__ Bt,
    float* __restrict__ C, int M, int N, int K) {
    __shared__ unsigned short As[2][TM][TK + 8];  // row stride 80B (64B data + 16B pad)
    __shared__ unsigned short Bs[2][TN][TK + 8];

    const int tid  = threadIdx.x;
    const int wave = tid >> 5;
    const int lane = tid & 31;
    const int col  = lane & 15;
    const int hi   = lane >> 4;
    const int m0 = blockIdx.y * TM;
    const int n0 = blockIdx.x * TN;

    v8f acc[2][4] = {};

    const int bRow = tid >> 1, bHalf = (tid & 1) * 16;

    auto stage = [&](int buf, int k0) {
        if (wave == 0) {   // whole A tile (128x32, padded rows) in one TDM descriptor
            tdm_load_2d((unsigned)(uintptr_t)&As[buf][0][0],
                        A + (size_t)m0 * K + k0,
                        K, M, TK, TM, K);
        }
        // B tile (already N x K): 2 threads per row, 32B each, async copies
        const unsigned short* bsrc = Bt + (size_t)(n0 + bRow) * K + k0 + bHalf;
        unsigned bdst = (unsigned)(uintptr_t)&Bs[buf][bRow][bHalf];
        async_load_b128(bdst,      bsrc);
        async_load_b128(bdst + 16, bsrc + 8);
    };
    auto stage_wait = [&]() {
        if (wave == 0) __builtin_amdgcn_s_wait_tensorcnt(0);
        wait_async0();
    };

    stage(0, 0);
    stage_wait();
    __syncthreads();

    int buf = 0;
    for (int k0 = 0; k0 < K; k0 += TK) {
        if (k0 + TK < K) stage(buf ^ 1, k0 + TK);   // overlap DMA with WMMA burst

        BF16Frag a0, a1, bf[4];
        {
            const int r = wave * 32 + col, kb = hi * 8;
            a0.u[0] = *(const u32x4*)&As[buf][r][kb];
            a0.u[1] = *(const u32x4*)&As[buf][r][16 + kb];
            a1.u[0] = *(const u32x4*)&As[buf][r + 16][kb];
            a1.u[1] = *(const u32x4*)&As[buf][r + 16][16 + kb];
        }
        #pragma unroll
        for (int t = 0; t < 4; ++t) {
            const int r = t * 16 + col, kb = hi * 8;
            bf[t].u[0] = *(const u32x4*)&Bs[buf][r][kb];
            bf[t].u[1] = *(const u32x4*)&Bs[buf][r][kb + 16];
        }
        #pragma unroll
        for (int t = 0; t < 4; ++t) {
            acc[0][t] = __builtin_amdgcn_wmma_f32_16x16x32_bf16(
                false, a0.v, false, bf[t].v, (short)0, acc[0][t], false, false);
            acc[1][t] = __builtin_amdgcn_wmma_f32_16x16x32_bf16(
                false, a1.v, false, bf[t].v, (short)0, acc[1][t], false, false);
        }
        stage_wait();
        __syncthreads();
        buf ^= 1;
    }
    #pragma unroll
    for (int i = 0; i < 2; ++i)
        #pragma unroll
        for (int t = 0; t < 4; ++t)
            #pragma unroll
            for (int v = 0; v < 8; ++v)
                C[(size_t)(m0 + wave * 32 + i * 16 + v + 8 * hi) * N +
                  n0 + t * 16 + col] = acc[i][t][v];
}

// ---------------- flash attention (causal, GQA) ----------------
// Q:(S,HQ,128) bf16 ; K:(S,HKV,128) bf16 ; Vt:(HKV*128, S) bf16 pre-transposed
__global__ __launch_bounds__(128) void flash_attn_wmma(
    const unsigned short* __restrict__ Q,
    const unsigned short* __restrict__ Kb,
    const unsigned short* __restrict__ Vtg,
    float* __restrict__ O) {
    const int h   = blockIdx.x;
    const int qb  = blockIdx.y;
    const int hkv = h >> 2;            // G = 4
    const int tid = threadIdx.x;
    const int wave = tid >> 5, lane = tid & 31;
    const int col = lane & 15, hi = lane >> 4;

    __shared__ unsigned short Klds[32][136];      // (key, d) row stride 272B
    __shared__ unsigned short Vt[128][40];        // (d, key) row stride 80B
    __shared__ float          Slds[4][16][32];
    __shared__ unsigned short Plds[4][16][40];
    __shared__ float          alpha_l[4][16];
    __shared__ float          rsum_l[4][16];

    const int r0 = qb * 64 + wave * 16;

    BF16Frag qf[4];
    {
        const unsigned short* qp = Q + ((size_t)(r0 + col) * HQ_N + h) * DH_N;
        const int kb = hi * 8;
        #pragma unroll
        for (int c = 0; c < 4; ++c) {
            qf[c].u[0] = *(const u32x4*)(qp + c * 32 + kb);
            qf[c].u[1] = *(const u32x4*)(qp + c * 32 + 16 + kb);
        }
    }

    v8f o[8] = {};
    float rmax = -__builtin_inff(), rsum = 0.0f;
    const float scale = 0.08838834764831845f;      // 1/sqrt(128)

    const int jend = qb * 64 + 64;
    for (int j0 = 0; j0 < jend; j0 += 32) {
        __syncthreads();
        {
            // K tile: async global->LDS, thread covers (key row, 1/4 of d)
            const int kr = tid >> 2, kc = (tid & 3) * 32;
            const unsigned short* ks = Kb + ((size_t)(j0 + kr) * HKV_N + hkv) * DH_N + kc;
            unsigned kd = (unsigned)(uintptr_t)&Klds[kr][kc];
            async_load_b128(kd,      ks);
            async_load_b128(kd + 16, ks + 8);
            async_load_b128(kd + 32, ks + 16);
            async_load_b128(kd + 48, ks + 24);
            // V tile from pre-transposed global: thread t stages d-row t (32 keys)
            const unsigned short* vs = Vtg + (size_t)(hkv * DH_N + tid) * S_LEN + j0;
            unsigned vd = (unsigned)(uintptr_t)&Vt[tid][0];
            async_load_b128(vd,      vs);
            async_load_b128(vd + 16, vs + 8);
            async_load_b128(vd + 32, vs + 16);
            async_load_b128(vd + 48, vs + 24);
        }
        wait_async0();
        __syncthreads();

        v8f s[2] = {};                          // S = Q * K^T (16 x 32)
        #pragma unroll
        for (int nt = 0; nt < 2; ++nt) {
            BF16Frag b[4];
            #pragma unroll
            for (int c = 0; c < 4; ++c) {
                const int r = nt * 16 + col, kb = c * 32 + hi * 8;
                b[c].u[0] = *(const u32x4*)&Klds[r][kb];
                b[c].u[1] = *(const u32x4*)&Klds[r][kb + 16];
            }
            #pragma unroll
            for (int c = 0; c < 4; ++c)
                s[nt] = __builtin_amdgcn_wmma_f32_16x16x32_bf16(
                    false, qf[c].v, false, b[c].v, (short)0, s[nt], false, false);
        }
        #pragma unroll
        for (int nt = 0; nt < 2; ++nt)
            #pragma unroll
            for (int v = 0; v < 8; ++v)
                Slds[wave][v + 8 * hi][nt * 16 + col] = s[nt][v];
        __syncthreads();

        if (lane < 16) {                        // online softmax, one row per lane
            const int qrow = r0 + lane;
            float m = rmax, sv[32];
            #pragma unroll
            for (int c = 0; c < 32; ++c) {
                float x = Slds[wave][lane][c] * scale;
                sv[c] = (j0 + c <= qrow) ? x : -__builtin_inff();
                m = fmaxf(m, sv[c]);
            }
            float a = __expf(rmax - m);
            float sum = rsum * a;
            #pragma unroll
            for (int c = 0; c < 32; ++c) {
                float p = __expf(sv[c] - m);
                sum += p;
                Plds[wave][lane][c] = f2bf(p);
            }
            rmax = m; rsum = sum;
            alpha_l[wave][lane] = a;
        }
        __syncthreads();

        #pragma unroll
        for (int v = 0; v < 8; ++v) {
            float a = alpha_l[wave][v + 8 * hi];
            #pragma unroll
            for (int t = 0; t < 8; ++t) o[t][v] *= a;
        }

        BF16Frag p;
        {
            const int kb = hi * 8;
            p.u[0] = *(const u32x4*)&Plds[wave][col][kb];
            p.u[1] = *(const u32x4*)&Plds[wave][col][16 + kb];
        }
        #pragma unroll
        for (int t = 0; t < 8; ++t) {           // O += P * V
            BF16Frag b;
            const int r = t * 16 + col, kb = hi * 8;
            b.u[0] = *(const u32x4*)&Vt[r][kb];
            b.u[1] = *(const u32x4*)&Vt[r][kb + 16];
            o[t] = __builtin_amdgcn_wmma_f32_16x16x32_bf16(
                false, p.v, false, b.v, (short)0, o[t], false, false);
        }
    }

    if (lane < 16) rsum_l[wave][lane] = rsum;
    __syncthreads();
    #pragma unroll
    for (int v = 0; v < 8; ++v) {
        float rinv = 1.0f / rsum_l[wave][v + 8 * hi];
        float* dst = O + (size_t)(r0 + v + 8 * hi) * DE_N + h * DH_N + col;
        #pragma unroll
        for (int t = 0; t < 8; ++t) dst[t * 16] = o[t][v] * rinv;
    }
}

// ---------------- host-side launcher ----------------
extern "C" void kernel_launch(void* const* d_in, const int* in_sizes, int n_in,
                              void* d_out, int out_size, void* d_ws, size_t ws_size,
                              hipStream_t stream) {
    const float* x  = (const float*)d_in[0];
    const float* cs = (const float*)d_in[1];
    const float* wq = (const float*)d_in[2];
    const float* wk = (const float*)d_in[3];
    const float* wv = (const float*)d_in[4];
    const float* wo = (const float*)d_in[5];
    float* out = (float*)d_out;

    char* ws = (char*)d_ws;
    const size_t MB = 1024 * 1024;
    unsigned short* xbf  = (unsigned short*)(ws + 0);        // 16 MB
    unsigned short* wqT  = (unsigned short*)(ws + 16 * MB);  // 32 MB (N x K)
    unsigned short* wkT  = (unsigned short*)(ws + 48 * MB);  //  8 MB
    unsigned short* wvT  = (unsigned short*)(ws + 56 * MB);  //  8 MB
    unsigned short* woT  = (unsigned short*)(ws + 64 * MB);  // 32 MB
    float*          Qf   = (float*)         (ws + 96 * MB);  // 32 MB
    float*          Kf   = (float*)         (ws + 128 * MB); //  8 MB
    float*          Vf   = (float*)         (ws + 136 * MB); //  8 MB
    unsigned short* Qr   = (unsigned short*)(ws + 144 * MB); // 16 MB
    unsigned short* Kr   = (unsigned short*)(ws + 160 * MB); //  4 MB
    unsigned short* Vtg  = (unsigned short*)(ws + 164 * MB); //  4 MB (HKV*128 x S)
    float*          Of   = (float*)         (ws + 168 * MB); // 32 MB
    unsigned short* Ob   = (unsigned short*)(ws + 200 * MB); // 16 MB

    // converts / weight transposes (one-time, bandwidth-bound)
    f32_to_bf16_kernel<<<2048, 256, 0, stream>>>(x, xbf, (long)S_LEN * DE_N);
    transpose_f32_to_bf16<<<dim3(DE_N / 32, DE_N / 32), 256, 0, stream>>>(
        wq, wqT, DE_N, DE_N);
    transpose_f32_to_bf16<<<dim3((HKV_N * DH_N) / 32, DE_N / 32), 256, 0, stream>>>(
        wk, wkT, DE_N, HKV_N * DH_N);
    transpose_f32_to_bf16<<<dim3((HKV_N * DH_N) / 32, DE_N / 32), 256, 0, stream>>>(
        wv, wvT, DE_N, HKV_N * DH_N);
    transpose_f32_to_bf16<<<dim3(DE_N / 32, DE_N / 32), 256, 0, stream>>>(
        wo, woT, DE_N, DE_N);

    // projections
    gemm_bf16_wmma<<<dim3(DE_N / TN, S_LEN / TM), 128, 0, stream>>>(
        xbf, wqT, Qf, S_LEN, DE_N, DE_N);
    gemm_bf16_wmma<<<dim3((HKV_N * DH_N) / TN, S_LEN / TM), 128, 0, stream>>>(
        xbf, wkT, Kf, S_LEN, HKV_N * DH_N, DE_N);
    gemm_bf16_wmma<<<dim3((HKV_N * DH_N) / TN, S_LEN / TM), 128, 0, stream>>>(
        xbf, wvT, Vf, S_LEN, HKV_N * DH_N, DE_N);

    // RoPE (Q,K) ; V transposed to (HKV*128, S)
    rope_bf16_kernel<<<(S_LEN * HQ_N * 64) / 256, 256, 0, stream>>>(Qf, cs, Qr, HQ_N);
    rope_bf16_kernel<<<(S_LEN * HKV_N * 64) / 256, 256, 0, stream>>>(Kf, cs, Kr, HKV_N);
    transpose_f32_to_bf16<<<dim3((HKV_N * DH_N) / 32, S_LEN / 32), 256, 0, stream>>>(
        Vf, Vtg, S_LEN, HKV_N * DH_N);

    // attention
    flash_attn_wmma<<<dim3(HQ_N, S_LEN / 64), 128, 0, stream>>>(Qr, Kr, Vtg, Of);

    // output projection
    f32_to_bf16_kernel<<<2048, 256, 0, stream>>>(Of, Ob, (long)S_LEN * DE_N);
    gemm_bf16_wmma<<<dim3(DE_N / TN, S_LEN / TM), 128, 0, stream>>>(
        Ob, woT, out, S_LEN, DE_N, DE_N);
}
